// MultiCrossEntropyModule_4037269258791
// MI455X (gfx1250) — compile-verified, matrix operand order checked by reading, not run on hardware
//
#include <hip/hip_runtime.h>
#include <math.h>

#define B_   4096
#define D_   1792
#define NC_  3000
#define NM_  7000
#define NF_  13000
#define RADF 0.017453292519943295f
#define REARTH 6371.0f

typedef __bf16 bf16_t;
typedef __attribute__((ext_vector_type(4)))  __bf16 v4bf;
typedef __attribute__((ext_vector_type(8)))  __bf16 v8bf;
typedef __attribute__((ext_vector_type(16))) __bf16 v16bf;
typedef __attribute__((ext_vector_type(8)))  float  v8f;

// ---------------------------------------------------------------- init
__global__ __launch_bounds__(32) void init_acc(float* acc) {
  if (threadIdx.x < 8) acc[threadIdx.x] = 0.0f;
}

// ----------------------------------------------- features f32 -> bf16
__global__ __launch_bounds__(256) void aconv_kernel(const float* __restrict__ A,
                                                    bf16_t* __restrict__ At) {
  size_t i = ((size_t)blockIdx.x * 256 + threadIdx.x) * 4;
  if (i >= (size_t)B_ * D_) return;
  float4 f = *(const float4*)(A + i);
  v4bf o;
  o[0] = (bf16_t)f.x; o[1] = (bf16_t)f.y; o[2] = (bf16_t)f.z; o[3] = (bf16_t)f.w;
  *(v4bf*)(At + i) = o;
}

// ------------------------------- weight transpose-convert: Wt[n][k] bf16
__global__ __launch_bounds__(256) void wtrans_kernel(const float* __restrict__ W,
                                                     bf16_t* __restrict__ Wt,
                                                     int N) {
  __shared__ bf16_t tile[64][72];   // padded to dodge bank conflicts
  const int tid = threadIdx.x;
  const int n0 = blockIdx.x * 64;
  const int k0 = blockIdx.y * 64;
  for (int idx = tid; idx < 64 * 64; idx += 256) {
    int nl = idx & 63, kl = idx >> 6;
    int n = n0 + nl; int nc = n < N ? n : (N - 1);
    tile[nl][kl] = (bf16_t)W[(size_t)(k0 + kl) * N + nc];
  }
  __syncthreads();
  for (int idx = tid; idx < 64 * 64; idx += 256) {
    int kl = idx & 63, nl = idx >> 6;
    int n = n0 + nl;
    if (n < N) Wt[(size_t)n * D_ + k0 + kl] = tile[nl][kl];
  }
}

// ------------------------------------------------------ partition trig
__global__ __launch_bounds__(256) void trig_kernel(const float* __restrict__ cp,
                                                   const float* __restrict__ mp,
                                                   const float* __restrict__ fp,
                                                   float* __restrict__ pt) {
  int i = blockIdx.x * 256 + threadIdx.x;
  const int total = NC_ + NM_ + NF_;
  if (i >= total) return;
  float lat, lon;
  if (i < NC_)            { lat = cp[2*i];            lon = cp[2*i+1]; }
  else if (i < NC_ + NM_) { int j = i - NC_;          lat = mp[2*j]; lon = mp[2*j+1]; }
  else                    { int j = i - NC_ - NM_;    lat = fp[2*j]; lon = fp[2*j+1]; }
  float la = lat * RADF, lo = lon * RADF;
  pt[4*i+0] = sinf(la); pt[4*i+1] = cosf(la);
  pt[4*i+2] = sinf(lo); pt[4*i+3] = cosf(lo);
}

// --------------------------------------------- grandparent index table
__global__ __launch_bounds__(256) void gp_kernel(const int* __restrict__ mpar,
                                                 const int* __restrict__ fpar,
                                                 int* __restrict__ gp) {
  int j = blockIdx.x * 256 + threadIdx.x;
  if (j < NF_) gp[j] = mpar[fpar[j]];
}

// ------------------------------------------------- label argmin kernel
__global__ __launch_bounds__(256) void label_kernel(const float* __restrict__ lab2d,
                                                    const float* __restrict__ pt,
                                                    int* __restrict__ labels) {
  __shared__ float sval[256];
  __shared__ int   sidx[256];
  const int b = blockIdx.x, tid = threadIdx.x;
  const float la = lab2d[2*b] * RADF, lo = lab2d[2*b+1] * RADF;
  const float sl = sinf(la), cl = cosf(la), so = sinf(lo), co = cosf(lo);
  const int off[3] = {0, NC_, NC_ + NM_};
  const int cnt[3] = {NC_, NM_, NF_};
  for (int h = 0; h < 3; ++h) {
    float best = INFINITY; int bi = 0;
    for (int i = tid; i < cnt[h]; i += 256) {
      const float* p = pt + 4 * (off[h] + i);
      float clcl  = cl * p[1];
      float cdlat = clcl + sl * p[0];
      float cdlon = co * p[3] + so * p[2];
      float hav   = 0.5f * (1.0f - cdlat) + clcl * 0.5f * (1.0f - cdlon);
      if (hav < best) { best = hav; bi = i; }
    }
    sval[tid] = best; sidx[tid] = bi;
    __syncthreads();
    for (int s = 128; s > 0; s >>= 1) {
      if (tid < s) {
        float ov = sval[tid + s]; int oi = sidx[tid + s];
        if (ov < sval[tid] || (ov == sval[tid] && oi < sidx[tid])) { sval[tid] = ov; sidx[tid] = oi; }
      }
      __syncthreads();
    }
    if (tid == 0) labels[h * B_ + b] = sidx[0];
    __syncthreads();
  }
}

// --------------------------------------------------------- WMMA GEMM
// C[B_ x N] = At[B_ x D_](bf16) x Wt[N x D_](bf16, K-major) + bias, f32 accum.
// One wave per 16(M) x 64(N) tile; K stepped by 32; software-pipelined
// (fragments for step k+1 are loaded before the WMMAs of step k execute).
static __device__ __forceinline__ v16bf load_frag(const bf16_t* __restrict__ p, int k) {
  v8bf lo = *(const v8bf*)(p + k);
  v8bf hi = *(const v8bf*)(p + k + 16);
  return __builtin_shufflevector(lo, hi, 0,1,2,3,4,5,6,7,8,9,10,11,12,13,14,15);
}

__global__ __launch_bounds__(256) void gemm_wmma(const bf16_t* __restrict__ At,
                                                 const bf16_t* __restrict__ Wt,
                                                 const float* __restrict__ bias,
                                                 float* __restrict__ C,
                                                 int N, int tilesN) {
  const int lane = threadIdx.x & 31;
  const int wave = threadIdx.x >> 5;
  const int t = blockIdx.x * 8 + wave;
  const int totalTiles = (B_ / 16) * tilesN;
  if (t >= totalTiles) return;
  const int mT = t / tilesN, nT = t % tilesN;
  const int m0 = mT * 16, n0 = nT * 64;
  const int hf = lane >> 4;         // half-wave select
  const int lr = lane & 15;

  const bf16_t* __restrict__ pA = At + (size_t)(m0 + lr) * D_ + 8 * hf;

  int col[4];
  const bf16_t* pB[4];
  #pragma unroll
  for (int s = 0; s < 4; ++s) {
    col[s] = n0 + 16 * s + lr;
    int cc = col[s] < N ? col[s] : (N - 1);
    pB[s] = Wt + (size_t)cc * D_ + 8 * hf;
  }

  v8f c0 = {}, c1 = {}, c2 = {}, c3 = {};

  // prologue: load fragments for k = 0
  v16bf a  = load_frag(pA, 0);
  v16bf b0 = load_frag(pB[0], 0);
  v16bf b1 = load_frag(pB[1], 0);
  v16bf b2 = load_frag(pB[2], 0);
  v16bf b3 = load_frag(pB[3], 0);

  for (int k0 = 0; k0 < D_ - 32; k0 += 32) {
    const int kn = k0 + 32;
    // issue next-step loads before consuming current fragments
    v16bf an  = load_frag(pA, kn);
    v16bf b0n = load_frag(pB[0], kn);
    v16bf b1n = load_frag(pB[1], kn);
    v16bf b2n = load_frag(pB[2], kn);
    v16bf b3n = load_frag(pB[3], kn);

    c0 = __builtin_amdgcn_wmma_f32_16x16x32_bf16(false, a, false, b0, (short)0, c0, false, false);
    c1 = __builtin_amdgcn_wmma_f32_16x16x32_bf16(false, a, false, b1, (short)0, c1, false, false);
    c2 = __builtin_amdgcn_wmma_f32_16x16x32_bf16(false, a, false, b2, (short)0, c2, false, false);
    c3 = __builtin_amdgcn_wmma_f32_16x16x32_bf16(false, a, false, b3, (short)0, c3, false, false);

    a = an; b0 = b0n; b1 = b1n; b2 = b2n; b3 = b3n;
  }
  // epilogue: last k-step
  c0 = __builtin_amdgcn_wmma_f32_16x16x32_bf16(false, a, false, b0, (short)0, c0, false, false);
  c1 = __builtin_amdgcn_wmma_f32_16x16x32_bf16(false, a, false, b1, (short)0, c1, false, false);
  c2 = __builtin_amdgcn_wmma_f32_16x16x32_bf16(false, a, false, b2, (short)0, c2, false, false);
  c3 = __builtin_amdgcn_wmma_f32_16x16x32_bf16(false, a, false, b3, (short)0, c3, false, false);

  // C layout: vgpr r -> row m0 + r + 8*hf, col = n0 + 16*s + lr
  #pragma unroll
  for (int s = 0; s < 4; ++s) {
    v8f acc = (s == 0) ? c0 : (s == 1) ? c1 : (s == 2) ? c2 : c3;
    if (col[s] < N) {
      float bv = bias[col[s]];
      #pragma unroll
      for (int r = 0; r < 8; ++r) {
        int row = m0 + r + 8 * hf;
        C[(size_t)row * N + col[s]] = acc[r] + bv;
      }
    }
  }
}

// ----------------------------------------------- fused NLL (online LSE)
__global__ __launch_bounds__(256) void loss_kernel(const float* __restrict__ Lc,
                                                   const float* __restrict__ Lm,
                                                   const float* __restrict__ Lf,
                                                   const int* __restrict__ labels,
                                                   float* __restrict__ acc) {
  __shared__ float sm[256];
  __shared__ float ss[256];
  const int b = blockIdx.x, tid = threadIdx.x;
  const float* Ls[3] = {Lc, Lm, Lf};
  const int    Ns[3] = {NC_, NM_, NF_};
  float nll_sum = 0.0f;
  for (int h = 0; h < 3; ++h) {
    const float* row = Ls[h] + (size_t)b * Ns[h];
    float m = -INFINITY, s = 0.0f;
    for (int j = tid; j < Ns[h]; j += 256) {
      float x = row[j];
      if (x > m) { s = s * expf(m - x) + 1.0f; m = x; }
      else       { s += expf(x - m); }
    }
    sm[tid] = m; ss[tid] = s;
    __syncthreads();
    for (int st = 128; st > 0; st >>= 1) {
      if (tid < st) {
        float m1 = sm[tid], s1 = ss[tid], m2 = sm[tid + st], s2 = ss[tid + st];
        float M = fmaxf(m1, m2);
        sm[tid] = M;
        ss[tid] = s1 * expf(m1 - M) + s2 * expf(m2 - M);
      }
      __syncthreads();
    }
    if (tid == 0) {
      float lse = sm[0] + logf(ss[0]);
      nll_sum += lse - row[labels[h * B_ + b]];
    }
    __syncthreads();
  }
  if (tid == 0) atomicAdd(&acc[0], nll_sum);
}

// ---------------------- combined-logit argmax prediction + accuracy
__global__ __launch_bounds__(256) void pred_kernel(const float* __restrict__ Lc,
                                                   const float* __restrict__ Lm,
                                                   const float* __restrict__ Lf,
                                                   const int* __restrict__ fpar,
                                                   const int* __restrict__ gp,
                                                   const float* __restrict__ fpart,
                                                   const float* __restrict__ lab2d,
                                                   float* __restrict__ acc) {
  __shared__ float sval[256];
  __shared__ int   sidx[256];
  const int b = blockIdx.x, tid = threadIdx.x;
  const float* rf = Lf + (size_t)b * NF_;
  const float* rm = Lm + (size_t)b * NM_;
  const float* rc = Lc + (size_t)b * NC_;
  float best = -INFINITY; int bi = 0;
  for (int j = tid; j < NF_; j += 256) {
    float s = rf[j] + rm[fpar[j]] + rc[gp[j]];
    if (s > best) { best = s; bi = j; }
  }
  sval[tid] = best; sidx[tid] = bi;
  __syncthreads();
  for (int st = 128; st > 0; st >>= 1) {
    if (tid < st) {
      float ov = sval[tid + st]; int oi = sidx[tid + st];
      if (ov > sval[tid] || (ov == sval[tid] && oi < sidx[tid])) { sval[tid] = ov; sidx[tid] = oi; }
    }
    __syncthreads();
  }
  if (tid == 0) {
    int j = sidx[0];
    float la1 = lab2d[2*b] * RADF, lo1 = lab2d[2*b+1] * RADF;
    float la2 = fpart[2*j] * RADF, lo2 = fpart[2*j+1] * RADF;
    float sdlat = sinf(0.5f * (la2 - la1));
    float sdlon = sinf(0.5f * (lo2 - lo1));
    float a = sdlat * sdlat + cosf(la1) * cosf(la2) * sdlon * sdlon;
    a = fminf(fmaxf(a, 0.0f), 1.0f);
    float d = 2.0f * REARTH * asinf(sqrtf(a));
    const float thr[5] = {1.0f, 25.0f, 200.0f, 750.0f, 2500.0f};
    #pragma unroll
    for (int t = 0; t < 5; ++t)
      if (d <= thr[t]) atomicAdd(&acc[1 + t], 1.0f);
  }
}

// ---------------------------------------------------------- finalize
__global__ __launch_bounds__(32) void finalize_kernel(const float* acc, float* out) {
  int t = threadIdx.x;
  if (t == 0) out[0] = acc[0] / (float)B_;
  else if (t <= 5) out[t] = acc[t] / (float)B_;
}

// ---------------------------------------------------------------- launch
extern "C" void kernel_launch(void* const* d_in, const int* in_sizes, int n_in,
                              void* d_out, int out_size, void* d_ws, size_t ws_size,
                              hipStream_t stream) {
  const float* feat   = (const float*)d_in[0];
  const float* lab2d  = (const float*)d_in[1];
  const float* cpart  = (const float*)d_in[2];
  const float* mpart  = (const float*)d_in[3];
  const float* fpart  = (const float*)d_in[4];
  const int*   mpar   = (const int*)d_in[5];
  const int*   fpar   = (const int*)d_in[6];
  const float* cw     = (const float*)d_in[7];
  const float* cb     = (const float*)d_in[8];
  const float* mw     = (const float*)d_in[9];
  const float* mb     = (const float*)d_in[10];
  const float* fw     = (const float*)d_in[11];
  const float* fb     = (const float*)d_in[12];
  float* out = (float*)d_out;

  // workspace layout
  float* Lc = (float*)d_ws;
  float* Lm = Lc + (size_t)B_ * NC_;
  float* Lf = Lm + (size_t)B_ * NM_;
  float* pt = Lf + (size_t)B_ * NF_;
  int* labels = (int*)(pt + 4 * (NC_ + NM_ + NF_));
  int* gp     = labels + 3 * B_;
  float* acc  = (float*)(gp + NF_);
  bf16_t* At  = (bf16_t*)(acc + 16);
  bf16_t* Wtc = At + (size_t)B_ * D_;
  bf16_t* Wtm = Wtc + (size_t)NC_ * D_;
  bf16_t* Wtf = Wtm + (size_t)NM_ * D_;

  init_acc<<<1, 32, 0, stream>>>(acc);

  // precision staging: bf16 A, transposed bf16 weights (K-major)
  {
    size_t n4 = ((size_t)B_ * D_) / 4;
    aconv_kernel<<<(int)((n4 + 255) / 256), 256, 0, stream>>>(feat, At);
  }
  {
    dim3 g((NC_ + 63) / 64, D_ / 64);
    wtrans_kernel<<<g, 256, 0, stream>>>(cw, Wtc, NC_);
  }
  {
    dim3 g((NM_ + 63) / 64, D_ / 64);
    wtrans_kernel<<<g, 256, 0, stream>>>(mw, Wtm, NM_);
  }
  {
    dim3 g((NF_ + 63) / 64, D_ / 64);
    wtrans_kernel<<<g, 256, 0, stream>>>(fw, Wtf, NF_);
  }

  {
    int total = NC_ + NM_ + NF_;
    trig_kernel<<<(total + 255) / 256, 256, 0, stream>>>(cpart, mpart, fpart, pt);
  }
  gp_kernel<<<(NF_ + 255) / 256, 256, 0, stream>>>(mpar, fpar, gp);
  label_kernel<<<B_, 256, 0, stream>>>(lab2d, pt, labels);

  // three GEMMs
  {
    int tilesN = (NC_ + 63) / 64;
    int waves  = (B_ / 16) * tilesN;
    gemm_wmma<<<(waves + 7) / 8, 256, 0, stream>>>(At, Wtc, cb, Lc, NC_, tilesN);
  }
  {
    int tilesN = (NM_ + 63) / 64;
    int waves  = (B_ / 16) * tilesN;
    gemm_wmma<<<(waves + 7) / 8, 256, 0, stream>>>(At, Wtm, mb, Lm, NM_, tilesN);
  }
  {
    int tilesN = (NF_ + 63) / 64;
    int waves  = (B_ / 16) * tilesN;
    gemm_wmma<<<(waves + 7) / 8, 256, 0, stream>>>(At, Wtf, fb, Lf, NF_, tilesN);
  }

  loss_kernel<<<B_, 256, 0, stream>>>(Lc, Lm, Lf, labels, acc);
  pred_kernel<<<B_, 256, 0, stream>>>(Lc, Lm, Lf, fpar, gp, fpart, lab2d, acc);
  finalize_kernel<<<1, 32, 0, stream>>>(acc, out);
}
